// TernaryLinearInference_45896020525661
// MI455X (gfx1250) — compile-verified, matrix operand order checked by reading, not run on hardware
//
#include <hip/hip_runtime.h>
#include <hip/hip_bf16.h>

// Ternary-packed linear layer: y[t][o] = sum_k x[t][k] * tern(w[o][k]) * scale[o] + bias[o]
// M=8192 (tokens), K=4096 (in), N=11008 (out). Compute-bound => bf16 WMMA path.
// Block tile: 256(M) x 128(N), K-step 32, double-buffered LDS, 8 wave32s,
// each wave computes 128x32 via 16 accumulators (16 v_wmma per K-stage).

typedef __attribute__((ext_vector_type(16))) __bf16 v16bf;
typedef __attribute__((ext_vector_type(8)))  __bf16 v8bf;
typedef __attribute__((ext_vector_type(8)))  float  v8f;

#define IN_F   4096
#define OUT_F  11008
#define TOKENS 8192
#define KT     32
#define MT_BLK 256
#define NT_BLK 128
#define LDS_S  40   // row stride in bf16 elems: 80B rows, 16B-aligned, conflict-free lane map

__device__ __forceinline__ unsigned int pack2bf(float a, float b) {
    // two f32 -> packed bf16 pair (round-to-nearest-even), pure 32-bit ALU ops
    unsigned int ua = __builtin_bit_cast(unsigned int, a);
    unsigned int ub = __builtin_bit_cast(unsigned int, b);
    ua += 0x7FFFu + ((ua >> 16) & 1u);
    ub += 0x7FFFu + ((ub >> 16) & 1u);
    return (ua >> 16) | (ub & 0xFFFF0000u);
}

struct V8Pair { v8bf lo; v8bf hi; };

__device__ __forceinline__ v16bf load_frag(const unsigned short (*T)[LDS_S], int row0, int lane) {
    // 16-bit A/B fragment layout (CDNA5 ISA 7.12.2):
    // lanes 0-15: row = lane,    VGPR0-3 hold K=0..7,  VGPR4-7 hold K=16..23
    // lanes16-31: row = lane-16, VGPR0-3 hold K=8..15, VGPR4-7 hold K=24..31
    const int r  = row0 + (lane & 15);
    const int kb = (lane >> 4) * 8;
    V8Pair p;
    p.lo = *(const v8bf*)&T[r][kb];
    p.hi = *(const v8bf*)&T[r][kb + 16];
    return __builtin_bit_cast(v16bf, p);
}

__global__ __launch_bounds__(256, 1)
void ternary_wmma_kernel(const float* __restrict__ x,
                         const unsigned char* __restrict__ wp,
                         const float* __restrict__ scale,
                         const float* __restrict__ bias,
                         float* __restrict__ y)
{
    __shared__ unsigned short Xs[2][MT_BLK][LDS_S];   // token rows x K-slice (bf16 bits)
    __shared__ unsigned short Ws[2][NT_BLK][LDS_S];   // out-feature rows x K-slice (bf16 bits)

    const int tid  = threadIdx.x;
    const int bm   = blockIdx.y;       // token tile (256 rows)
    const int bn   = blockIdx.x;       // out-feature tile (128 cols)
    const int lane = tid & 31;
    const int wv   = tid >> 5;         // 8 waves
    const int wv_m = wv & 1;           // 2 M-groups of 128
    const int wv_n = wv >> 1;          // 4 N-groups of 32

    v8f acc[8][2];
    #pragma unroll
    for (int mt = 0; mt < 8; ++mt)
        #pragma unroll
        for (int nt = 0; nt < 2; ++nt)
            acc[mt][nt] = (v8f){0.f,0.f,0.f,0.f,0.f,0.f,0.f,0.f};

    auto load_stage = [&](int kk, int buf) {
        // --- X tile: 256 x 32 f32 -> bf16. Each thread: 8 float4 loads. ---
        {
            const int r0 = tid >> 3;            // 0..31
            const int c4 = (tid & 7) * 4;       // 0,4,...,28
            #pragma unroll
            for (int i = 0; i < 8; ++i) {
                const int r = r0 + 32 * i;
                const float4 v = *(const float4*)(x + (size_t)(bm * MT_BLK + r) * IN_F + kk + c4);
                uint2 pk;
                pk.x = pack2bf(v.x, v.y);
                pk.y = pack2bf(v.z, v.w);
                *(uint2*)&Xs[buf][r][c4] = pk;
            }
            // speculative prefetch of the K-slice after next (CDNA5 global_prefetch)
            if (kk + 2 * KT < IN_F)
                __builtin_prefetch(x + (size_t)(bm * MT_BLK + r0) * IN_F + kk + 2 * KT, 0, 1);
        }
        // --- W tile: 128 rows x 8 packed bytes -> 128 x 32 bf16. Each thread: one u32 = 16 codes. ---
        {
            const int n = tid >> 1;
            const int q = tid & 1;
            const unsigned int w32 =
                *(const unsigned int*)(wp + (size_t)(bn * NT_BLK + n) * (IN_F / 4) + (kk >> 2) + q * 4);
            // code 0 -> -1.0 (0xBF80), 1 -> 0.0, 2 -> +1.0 (0x3F80) in bf16 bits
            const unsigned long long LUT = (0x3F80ull << 32) | 0xBF80ull;
            unsigned int tmp[8];
            #pragma unroll
            for (int j = 0; j < 8; ++j) {
                const unsigned int c0 = (w32 >> (4 * j))     & 3u;
                const unsigned int c1 = (w32 >> (4 * j + 2)) & 3u;
                tmp[j] = (unsigned int)(unsigned short)(LUT >> (c0 * 16))
                       | ((unsigned int)(unsigned short)(LUT >> (c1 * 16)) << 16);
            }
            *(uint4*)&Ws[buf][n][q * 16]     = *(const uint4*)&tmp[0];
            *(uint4*)&Ws[buf][n][q * 16 + 8] = *(const uint4*)&tmp[4];
        }
    };

    load_stage(0, 0);
    __syncthreads();

    for (int kk = 0; kk < IN_F; kk += KT) {
        const int buf = (kk >> 5) & 1;
        if (kk + KT < IN_F)
            load_stage(kk + KT, buf ^ 1);   // fill other buffer while computing this one

        v16bf bfrag[2];
        #pragma unroll
        for (int nt = 0; nt < 2; ++nt)
            bfrag[nt] = load_frag(Ws[buf], wv_n * 32 + nt * 16, lane);

        #pragma unroll
        for (int mt = 0; mt < 8; ++mt) {
            const v16bf afrag = load_frag(Xs[buf], wv_m * 128 + mt * 16, lane);
            #pragma unroll
            for (int nt = 0; nt < 2; ++nt)
                acc[mt][nt] = __builtin_amdgcn_wmma_f32_16x16x32_bf16(
                    false, afrag, false, bfrag[nt], (short)0, acc[mt][nt], false, false);
        }
        __syncthreads();
    }

    // Epilogue: C/D layout = lane L: N = L%16; VGPR v: M = v + 8*(L>=16)
    const int nl = lane & 15;
    const int mh = (lane >> 4) * 8;
    #pragma unroll
    for (int nt = 0; nt < 2; ++nt) {
        const int n_g = bn * NT_BLK + wv_n * 32 + nt * 16 + nl;
        const float s = scale[n_g];
        const float b = bias[n_g];
        #pragma unroll
        for (int mt = 0; mt < 8; ++mt) {
            const int m_base = bm * MT_BLK + wv_m * 128 + mt * 16 + mh;
            #pragma unroll
            for (int v = 0; v < 8; ++v)
                y[(size_t)(m_base + v) * OUT_F + n_g] = acc[mt][nt][v] * s + b;
        }
    }
}

extern "C" void kernel_launch(void* const* d_in, const int* in_sizes, int n_in,
                              void* d_out, int out_size, void* d_ws, size_t ws_size,
                              hipStream_t stream) {
    const float*         x     = (const float*)d_in[0];
    const unsigned char* wp    = (const unsigned char*)d_in[1];
    const float*         scale = (const float*)d_in[2];
    const float*         bias  = (const float*)d_in[3];
    float*               y     = (float*)d_out;
    (void)in_sizes; (void)n_in; (void)out_size; (void)d_ws; (void)ws_size;

    dim3 grid(OUT_F / NT_BLK, TOKENS / MT_BLK);   // 86 x 32 workgroups
    ternary_wmma_kernel<<<grid, dim3(256), 0, stream>>>(x, wp, scale, bias, y);
}
